// TwoLayerSAGE_78520592106143
// MI455X (gfx1250) — matrix operand order repaired and checked by reading, not compiled, not run on hardware
//
#include <hip/hip_runtime.h>
#include <hip/hip_bf16.h>

// ---------------------------------------------------------------------------
// TwoLayerSAGE on MI455X (gfx1250): all four GEMMs on v_wmma_f32_16x16x32_f16.
// Aggregation GEMM: wave computes a 32x64 macro-tile (MT=2 x NT=4 accumulators)
// with a ping-pong (unroll-by-2) k-loop -> 8 WMMAs per buffer set, ~0.75KB
// loaded per WMMA, VMEM latency covered by a full iteration of matrix work.
// ---------------------------------------------------------------------------

typedef __attribute__((ext_vector_type(16))) _Float16 v16h;
typedef __attribute__((ext_vector_type(8)))  _Float16 v8h;
typedef __attribute__((ext_vector_type(8)))  float    v8f;

static constexpr int BB    = 8;
static constexpr int NN    = 2048;
static constexpr int IN_C  = 128;
static constexpr int HID_C = 256;
static constexpr int OUT_C = 32;

// ---- WMMA operand loaders -------------------------------------------------
// A-matrix 16x32 f16 (M x K), row-major source with leading dim ld (halves).
// lane L: M = L&15; lanes 0-15 hold K{0..7,16..23}, lanes 16-31 K{8..15,24..31}
// (ISA 7.12.2, 16-bit A layout).
__device__ __forceinline__ v16h load_a_tile(const _Float16* __restrict__ base,
                                            int ld, int lane) {
  int m  = lane & 15;
  int ko = (lane >> 4) << 3;                 // 0 or 8
  const _Float16* p = base + (size_t)m * ld + ko;
  v8h lo = *(const v8h*)(p);                 // K = ko .. ko+7
  v8h hi = *(const v8h*)(p + 16);            // K = ko+16 .. ko+23
  v16h r;
#pragma unroll
  for (int i = 0; i < 8; ++i) { r[i] = lo[i]; r[i + 8] = hi[i]; }
  return r;
}

// B-matrix 32x16 f16 (K x N), row-major source: lane L holds column N=L&15,
// K{0..7,16..23} (lanes 0-15) or K{8..15,24..31} (lanes 16-31).
__device__ __forceinline__ v16h load_b_tile(const _Float16* __restrict__ base,
                                            int ld, int lane) {
  int n  = lane & 15;
  int ko = (lane >> 4) << 3;
  v16h r;
#pragma unroll
  for (int i = 0; i < 8; ++i) {
    r[i]     = base[(size_t)(ko + i) * ld + n];
    r[i + 8] = base[(size_t)(ko + 16 + i) * ld + n];
  }
  return r;
}

__device__ __forceinline__ v8f wmma_f16(v16h a, v16h b, v8f c) {
  return __builtin_amdgcn_wmma_f32_16x16x32_f16(false, a, false, b,
                                                (short)0, c, false, false);
}

// ---- mask transpose + degree (one pass over A, coalesced both ways) -------
__global__ __launch_bounds__(256)
void mask_transpose_deg(const int* __restrict__ A, _Float16* __restrict__ maskT,
                        float* __restrict__ deg, int N) {
  __shared__ _Float16 tile[32][33];
  __shared__ float    csum[8][32];
  int tx = threadIdx.x, ty = threadIdx.y;      // 32 x 8
  int j0 = blockIdx.x * 32;
  int i0 = blockIdx.y * 32;
  int b  = blockIdx.z;
  const int* Ab = A + (size_t)b * N * N;
  float part = 0.f;
#pragma unroll
  for (int r = 0; r < 4; ++r) {
    int i = i0 + ty + r * 8;
    int a = Ab[(size_t)i * N + j0 + tx];
    float m = (a != 0) ? 1.f : 0.f;
    tile[ty + r * 8][tx] = (_Float16)m;
    part += m;                                  // partial of deg[b][j0+tx]
  }
  csum[ty][tx] = part;
  __syncthreads();
  if (ty == 0) {
    float s = 0.f;
#pragma unroll
    for (int r = 0; r < 8; ++r) s += csum[r][tx];
    atomicAdd(&deg[(size_t)b * N + j0 + tx], s);
  }
  _Float16* Mt = maskT + (size_t)b * N * N;
#pragma unroll
  for (int r = 0; r < 4; ++r) {
    int j = j0 + ty + r * 8;
    Mt[(size_t)j * N + i0 + tx] = tile[tx][ty + r * 8];
  }
}

__global__ void deg_finalize(float* deg, int n) {
  int i = blockIdx.x * blockDim.x + threadIdx.x;
  if (i < n) deg[i] = 1.f / fmaxf(deg[i], 1.f);
}

__global__ void f32_to_f16(const float* __restrict__ in,
                           _Float16* __restrict__ out, int n) {
  int i = blockIdx.x * blockDim.x + threadIdx.x;
  if (i < n) out[i] = (_Float16)in[i];
}

// ---- aggregation GEMM: out[b,m,:] = deg_inv[b,m] * (maskT[b] @ X[b]) ------
// Wave computes (16*MT) rows x (16*NT) cols; ping-pong k-loop (K % 64 == 0).
template <int MT, int NT>
__global__ __launch_bounds__(32)
void agg_gemm_f16(const _Float16* __restrict__ maskT,   // [B][M][K]
                  const _Float16* __restrict__ X,       // [B][K][D]
                  const float*    __restrict__ deg_inv, // [B][M]
                  _Float16*       __restrict__ out,     // [B][M][D]
                  int M, int K, int D) {
  int lane = threadIdx.x;
  int n0 = blockIdx.x * (16 * NT);
  int m0 = blockIdx.y * (16 * MT);
  int b  = blockIdx.z;
  const _Float16* Abase = maskT + (size_t)b * M * K + (size_t)m0 * K;
  const _Float16* Bbase = X + (size_t)b * K * D + n0;

  v8f acc[MT][NT] = {};
  // prologue: fill both buffer sets (k = 0 and k = 32)
  v16h a0[MT], a1[MT], b0[NT], b1[NT];
#pragma unroll
  for (int i = 0; i < MT; ++i)
    a0[i] = load_a_tile(Abase + (size_t)i * 16 * K, K, lane);
#pragma unroll
  for (int j = 0; j < NT; ++j) b0[j] = load_b_tile(Bbase + j * 16, D, lane);
#pragma unroll
  for (int i = 0; i < MT; ++i)
    a1[i] = load_a_tile(Abase + (size_t)i * 16 * K + 32, K, lane);
#pragma unroll
  for (int j = 0; j < NT; ++j)
    b1[j] = load_b_tile(Bbase + (size_t)32 * D + j * 16, D, lane);

  for (int k = 64; k + 64 <= K; k += 64) {
    __builtin_prefetch(Bbase + (size_t)(k + 64) * D, 0, 1);
    // consume buffer 0 (data for k-64), refill it with data for k
#pragma unroll
    for (int i = 0; i < MT; ++i)
#pragma unroll
      for (int j = 0; j < NT; ++j)
        acc[i][j] = wmma_f16(a0[i], b0[j], acc[i][j]);
#pragma unroll
    for (int i = 0; i < MT; ++i)
      a0[i] = load_a_tile(Abase + (size_t)i * 16 * K + k, K, lane);
#pragma unroll
    for (int j = 0; j < NT; ++j)
      b0[j] = load_b_tile(Bbase + (size_t)k * D + j * 16, D, lane);
    // consume buffer 1 (data for k-32), refill it with data for k+32
#pragma unroll
    for (int i = 0; i < MT; ++i)
#pragma unroll
      for (int j = 0; j < NT; ++j)
        acc[i][j] = wmma_f16(a1[i], b1[j], acc[i][j]);
#pragma unroll
    for (int i = 0; i < MT; ++i)
      a1[i] = load_a_tile(Abase + (size_t)i * 16 * K + k + 32, K, lane);
#pragma unroll
    for (int j = 0; j < NT; ++j)
      b1[j] = load_b_tile(Bbase + (size_t)(k + 32) * D + j * 16, D, lane);
  }
  // drain both buffers (last two k-steps)
#pragma unroll
  for (int i = 0; i < MT; ++i)
#pragma unroll
    for (int j = 0; j < NT; ++j) acc[i][j] = wmma_f16(a0[i], b0[j], acc[i][j]);
#pragma unroll
  for (int i = 0; i < MT; ++i)
#pragma unroll
    for (int j = 0; j < NT; ++j) acc[i][j] = wmma_f16(a1[i], b1[j], acc[i][j]);

  // epilogue: per-row deg_inv scale, f16 store
  int n  = lane & 15;
  int mo = (lane >> 4) << 3;
#pragma unroll
  for (int i = 0; i < MT; ++i) {
    _Float16* obase =
        out + (size_t)b * M * D + (size_t)(m0 + i * 16) * D + n0;
#pragma unroll
    for (int v = 0; v < 8; ++v) {
      float s = deg_inv[(size_t)b * M + m0 + i * 16 + mo + v];
#pragma unroll
      for (int j = 0; j < NT; ++j)
        obase[(size_t)(mo + v) * D + j * 16 + n] = (_Float16)(acc[i][j][v] * s);
    }
  }
}

// ---- fused dual GEMM: A1@W1 + A2@W2 + bias (+ReLU), f16 or f32 output -----
template <int NT, bool RELU, bool OUT16>
__global__ __launch_bounds__(32)
void dual_gemm_bias(const _Float16* __restrict__ A1,   // [B][M][K]
                    const _Float16* __restrict__ W1,   // [K][D]
                    const _Float16* __restrict__ A2,   // [B][M][K]
                    const _Float16* __restrict__ W2,   // [K][D]
                    const float*    __restrict__ bias, // [D]
                    _Float16* __restrict__ out16, float* __restrict__ out32,
                    int M, int K, int D) {
  int lane = threadIdx.x;
  int n0 = blockIdx.x * (16 * NT);
  int m0 = blockIdx.y * 16;
  int b  = blockIdx.z;
  const _Float16* a1 = A1 + (size_t)b * M * K + (size_t)m0 * K;
  const _Float16* a2 = A2 + (size_t)b * M * K + (size_t)m0 * K;

  v8f acc[NT] = {};
  for (int k = 0; k < K; k += 32) {
    // gather all operands for this k-step first (one clause), then 2*NT WMMAs
    v16h av1 = load_a_tile(a1 + k, K, lane);
    v16h av2 = load_a_tile(a2 + k, K, lane);
    v16h w1[NT], w2[NT];
#pragma unroll
    for (int j = 0; j < NT; ++j) {
      w1[j] = load_b_tile(W1 + (size_t)k * D + n0 + j * 16, D, lane);
      w2[j] = load_b_tile(W2 + (size_t)k * D + n0 + j * 16, D, lane);
    }
#pragma unroll
    for (int j = 0; j < NT; ++j) acc[j] = wmma_f16(av1, w1[j], acc[j]);
#pragma unroll
    for (int j = 0; j < NT; ++j) acc[j] = wmma_f16(av2, w2[j], acc[j]);
  }

  int n  = lane & 15;
  int mo = (lane >> 4) << 3;
#pragma unroll
  for (int j = 0; j < NT; ++j) {
    float bv = bias[n0 + j * 16 + n];
#pragma unroll
    for (int v = 0; v < 8; ++v) {
      float r = acc[j][v] + bv;
      if (RELU) r = fmaxf(r, 0.f);
      size_t idx =
          (size_t)b * M * D + (size_t)(m0 + mo + v) * D + n0 + j * 16 + n;
      if (OUT16) out16[idx] = (_Float16)r;
      else       out32[idx] = r;
    }
  }
}

// ---- log_softmax over last dim (=32) : one wave32 per row -----------------
__global__ __launch_bounds__(256)
void log_softmax32(const float* __restrict__ logits, float* __restrict__ out,
                   int rows) {
  int lane = threadIdx.x & 31;
  int wave = (int)((blockIdx.x * blockDim.x + threadIdx.x) >> 5);
  if (wave >= rows) return;
  float v = logits[(size_t)wave * 32 + lane];
  float m = v;
#pragma unroll
  for (int o = 16; o > 0; o >>= 1) m = fmaxf(m, __shfl_xor(m, o, 32));
  float e = __expf(v - m);
  float s = e;
#pragma unroll
  for (int o = 16; o > 0; o >>= 1) s += __shfl_xor(s, o, 32);
  out[(size_t)wave * 32 + lane] = (v - m) - __logf(s);
}

// ---------------------------------------------------------------------------
extern "C" void kernel_launch(void* const* d_in, const int* in_sizes, int n_in,
                              void* d_out, int out_size, void* d_ws, size_t ws_size,
                              hipStream_t stream) {
  const float* x    = (const float*)d_in[0];   // [B][N][IN_C]
  const int*   A    = (const int*)  d_in[1];   // [B][N][N]
  const float* W1l  = (const float*)d_in[2];   // [IN_C][HID_C]
  const float* W1r  = (const float*)d_in[3];
  const float* b1   = (const float*)d_in[4];   // [HID_C]
  const float* W2l  = (const float*)d_in[5];   // [HID_C][OUT_C]
  const float* W2r  = (const float*)d_in[6];
  const float* b2   = (const float*)d_in[7];   // [OUT_C]

  float* out_lsm    = (float*)d_out;                              // tuple[0]
  float* out_logits = (float*)d_out + (size_t)BB * NN * OUT_C;    // tuple[1]

  // workspace carve-up (256B aligned regions)
  char* ws = (char*)d_ws;
  auto carve = [&](size_t bytes) {
    char* p = ws;
    ws += (bytes + 255) & ~(size_t)255;
    return p;
  };
  _Float16* maskT  = (_Float16*)carve((size_t)BB * NN * NN * 2);      // 64 MiB
  _Float16* x16    = (_Float16*)carve((size_t)BB * NN * IN_C * 2);    //  4 MiB
  _Float16* agg1   = (_Float16*)carve((size_t)BB * NN * IN_C * 2);    //  4 MiB
  _Float16* h16    = (_Float16*)carve((size_t)BB * NN * HID_C * 2);   //  8 MiB
  _Float16* aggh   = (_Float16*)carve((size_t)BB * NN * HID_C * 2);   //  8 MiB
  _Float16* w1l16  = (_Float16*)carve((size_t)IN_C * HID_C * 2);
  _Float16* w1r16  = (_Float16*)carve((size_t)IN_C * HID_C * 2);
  _Float16* w2l16  = (_Float16*)carve((size_t)HID_C * OUT_C * 2);
  _Float16* w2r16  = (_Float16*)carve((size_t)HID_C * OUT_C * 2);
  float*    deg    = (float*)   carve((size_t)BB * NN * 4);           // deg -> deg_inv

  // 1) degree accumulator must start at zero
  hipMemsetAsync(deg, 0, (size_t)BB * NN * sizeof(float), stream);

  // 2) one pass over A: int32 -> f16 transpose + column degrees
  mask_transpose_deg<<<dim3(NN / 32, NN / 32, BB), dim3(32, 8), 0, stream>>>(
      A, maskT, deg, NN);
  deg_finalize<<<(BB * NN + 255) / 256, 256, 0, stream>>>(deg, BB * NN);

  // 3) f32 -> f16 conversions
  {
    int n = BB * NN * IN_C;
    f32_to_f16<<<(n + 255) / 256, 256, 0, stream>>>(x, x16, n);
    n = IN_C * HID_C;
    f32_to_f16<<<(n + 255) / 256, 256, 0, stream>>>(W1l, w1l16, n);
    f32_to_f16<<<(n + 255) / 256, 256, 0, stream>>>(W1r, w1r16, n);
    n = HID_C * OUT_C;
    f32_to_f16<<<(n + 255) / 256, 256, 0, stream>>>(W2l, w2l16, n);
    f32_to_f16<<<(n + 255) / 256, 256, 0, stream>>>(W2r, w2r16, n);
  }

  // 4) layer-1 aggregation: agg1 = deg_inv * (maskT @ x)   [M=2048,K=2048,D=128]
  agg_gemm_f16<2, 4><<<dim3(IN_C / 64, NN / 32, BB), 32, 0, stream>>>(
      maskT, x16, deg, agg1, NN, NN, IN_C);

  // 5) h = relu(agg1@W1_l + x@W1_r + b1)                   [K=128, D=256]
  dual_gemm_bias<4, true, true><<<dim3(HID_C / 64, NN / 16, BB), 32, 0, stream>>>(
      agg1, w1l16, x16, w1r16, b1, h16, (float*)nullptr, NN, IN_C, HID_C);

  // 6) layer-2 aggregation: aggh = deg_inv * (maskT @ h)   [K=2048, D=256]
  agg_gemm_f16<2, 4><<<dim3(HID_C / 64, NN / 32, BB), 32, 0, stream>>>(
      maskT, h16, deg, aggh, NN, NN, HID_C);

  // 7) logits = aggh@W2_l + h@W2_r + b2  -> tuple[1] (f32) [K=256, D=32]
  dual_gemm_bias<2, false, false><<<dim3(OUT_C / 32, NN / 16, BB), 32, 0, stream>>>(
      aggh, w2l16, h16, w2r16, b2, (_Float16*)nullptr, out_logits, NN, HID_C, OUT_C);

  // 8) log_softmax over last dim -> tuple[0]
  {
    int rows = BB * NN;
    log_softmax32<<<(rows * 32 + 255) / 256, 256, 0, stream>>>(out_logits,
                                                               out_lsm, rows);
  }
}
